// Mesh_Reduced_49435073577190
// MI455X (gfx1250) — matrix operand order repaired and checked by reading, not compiled, not run on hardware
//
#include <hip/hip_runtime.h>
#include <hip/hip_bf16.h>

typedef __bf16 v16bf __attribute__((ext_vector_type(16)));
typedef float  v8f   __attribute__((ext_vector_type(8)));
typedef int    v4i   __attribute__((vector_size(16)));  // matches builtin proto

#define LN_EPS 1e-5f

// ---------------------------------------------------------------------
// CDNA5 async global->LDS copy (16B per lane), ASYNCcnt-tracked.
// Builtin signature (from clang diagnostic): param0 = int4 AS1*, so:
//   (global int4*, lds int4*, imm offset, imm cpol)
// ---------------------------------------------------------------------
__device__ __forceinline__ void async_copy_b128(void* lds_dst, const void* gsrc) {
#if __has_builtin(__builtin_amdgcn_global_load_async_to_lds_b128)
  __builtin_amdgcn_global_load_async_to_lds_b128(
      (__attribute__((address_space(1))) v4i*)gsrc,
      (__attribute__((address_space(3))) v4i*)lds_dst, 0, 0);
#else
  *(uint4*)lds_dst = *(const uint4*)gsrc;
#endif
}

__device__ __forceinline__ void wait_async0() {
#if __has_builtin(__builtin_amdgcn_s_wait_asynccnt)
  __builtin_amdgcn_s_wait_asynccnt(0);
#else
  asm volatile("s_wait_asynccnt 0" ::: "memory");
#endif
}

// =====================================================================
// WMMA core with fragment-ordered packed bf16 weights.
// Wp layout: [(ks*8 + ntile)*32 + lane] * 16 bf16 -> one v16bf per lane.
// =====================================================================
template<int K>
__device__ __forceinline__ void wave_gemm_packed(const __bf16* __restrict__ A,
                                                 const __bf16* __restrict__ Wp,
                                                 int lane, int mstrip, int nbase,
                                                 v8f acc[4]) {
  const int nt0 = nbase >> 4;
#pragma unroll 1
  for (int ks = 0; ks < K / 32; ++ks) {
    // ---- A fragment: 16x32 bf16 per ISA layout (2x ds_load_b128) ----
    v16bf a;
    const int m  = mstrip + (lane & 15);
    const int hf = lane >> 4;
    const __bf16* ar = A + m * K + ks * 32;
#pragma unroll
    for (int r = 0; r < 4; ++r) {
      const int kk = hf * 8 + 2 * r;
      a[2 * r]         = ar[kk];
      a[2 * r + 1]     = ar[kk + 1];
      a[8 + 2 * r]     = ar[16 + kk];
      a[8 + 2 * r + 1] = ar[16 + kk + 1];
    }
#pragma unroll
    for (int t = 0; t < 4; ++t) {
      // ---- B fragment: one contiguous 32B load per lane ----
      const v16bf b = *(const v16bf*)(Wp +
          ((size_t)((ks * 8 + nt0 + t) * 32 + lane) << 4));
      acc[t] = __builtin_amdgcn_wmma_f32_16x16x32_bf16(
          false, a, false, b, (short)0, acc[t], false, false);
    }
  }
}

// =====================================================================
// Fused MGN block MLP:  out = LN(relu(A@W1+b1)@W2+b2)*g+b + resid
// KD==384: edge block, A row = [ef[e] | nf[src[e]] | nf[dst[e]]] (all bf16,
//          gathered via async global->LDS byte copies)
// KD==256: node block, A row = [agg[n]*invdeg[n] (fp32->bf16) | nf[n] (async)]
// 256 threads (8 waves), 64 rows x 128 cols per workgroup. 64KB LDS.
// =====================================================================
template<int KD>
__global__ __launch_bounds__(256) void mgn_block_kernel(
    const __bf16* __restrict__ xin,     // ef (E x 128) bf16, edge mode
    const float* __restrict__ agg,      // agg (N x 128) fp32, node mode
    const float* __restrict__ invdeg,   // node mode only
    const __bf16* __restrict__ nf,      // node features N x 128 bf16
    const int* __restrict__ src, const int* __restrict__ dst,
    const __bf16* __restrict__ W1p, const float* __restrict__ b1,
    const __bf16* __restrict__ W2p, const float* __restrict__ b2,
    const float* __restrict__ lng, const float* __restrict__ lnb,
    const __bf16* __restrict__ resid,
    __bf16* __restrict__ out, int M) {
  __shared__ char smem[65536];
  __bf16* sA   = (__bf16*)smem;                // 64*KD bf16  (<=48KB)
  __bf16* sH   = (__bf16*)(smem + 49152);      // 64*128 bf16 (16KB)
  float*  sOut = (float*)smem;                 // overlays sA after stage 1
  float*  sRed = (float*)(smem + 49152);       // overlays sH after stage 2
  int*    sSrc = (int*)(smem + 49152);         // during gather only
  int*    sDst = (int*)(smem + 49152 + 256);

  const int m0   = blockIdx.x * 64;
  const int tid  = threadIdx.x;
  const int lane = tid & 31;
  const int wave = tid >> 5;
  const int mstrip = (wave & 3) * 16;
  const int nbase  = (wave >> 2) * 64;

  if (KD == 384 && tid < 64) {
    const int gm = m0 + tid;
    const int gr = (gm < M) ? gm : (M - 1);
    sSrc[tid] = src[gr];
    sDst[tid] = dst[gr];
  }
  __syncthreads();

  // ---- gather A tile: bf16 byte copies (async) + agg conversion ----
  const int CPR = KD / 8;  // 16-byte chunks per row
  for (int i = tid; i < 64 * CPR; i += 256) {
    const int row = i / CPR;
    const int cc  = (i % CPR) * 8;     // bf16 column
    const int gm  = m0 + row;
    const int gr  = (gm < M) ? gm : (M - 1);
    if (KD == 384) {
      const __bf16* gp;
      if (cc < 128)      gp = xin + (size_t)gr * 128 + cc;
      else if (cc < 256) gp = nf + (size_t)sSrc[row] * 128 + (cc - 128);
      else               gp = nf + (size_t)sDst[row] * 128 + (cc - 256);
      async_copy_b128(sA + (size_t)row * KD + cc, gp);
    } else {
      if (cc < 128) {
        const float4 v0 = *(const float4*)(agg + (size_t)gr * 128 + cc);
        const float4 v1 = *(const float4*)(agg + (size_t)gr * 128 + cc + 4);
        const float id = invdeg[gr];
        __bf16* d = sA + (size_t)row * KD + cc;
        d[0] = (__bf16)(v0.x * id); d[1] = (__bf16)(v0.y * id);
        d[2] = (__bf16)(v0.z * id); d[3] = (__bf16)(v0.w * id);
        d[4] = (__bf16)(v1.x * id); d[5] = (__bf16)(v1.y * id);
        d[6] = (__bf16)(v1.z * id); d[7] = (__bf16)(v1.w * id);
      } else {
        async_copy_b128(sA + (size_t)row * KD + cc,
                        nf + (size_t)gr * 128 + (cc - 128));
      }
    }
  }
  wait_async0();
  __syncthreads();

  // ---- stage 1: h = relu(A @ W1 + b1) -> bf16 LDS ----
  {
    v8f acc[4] = {};
    wave_gemm_packed<KD>(sA, W1p, lane, mstrip, nbase, acc);
#pragma unroll
    for (int t = 0; t < 4; ++t) {
      const int col = nbase + t * 16 + (lane & 15);
      const float bb = b1[col];
#pragma unroll
      for (int r = 0; r < 8; ++r) {
        const int row = mstrip + (lane >> 4) * 8 + r;
        sH[row * 128 + col] = (__bf16)fmaxf(acc[t][r] + bb, 0.f);
      }
    }
  }
  __syncthreads();

  // ---- stage 2: o = h @ W2 + b2 -> fp32 LDS (overlays A) ----
  {
    v8f acc[4] = {};
    wave_gemm_packed<128>(sH, W2p, lane, mstrip, nbase, acc);
#pragma unroll
    for (int t = 0; t < 4; ++t) {
      const int col = nbase + t * 16 + (lane & 15);
      const float bb = b2[col];
#pragma unroll
      for (int r = 0; r < 8; ++r) {
        const int row = mstrip + (lane >> 4) * 8 + r;
        sOut[row * 128 + col] = acc[t][r] + bb;
      }
    }
  }
  __syncthreads();

  // ---- LayerNorm + residual, 4 threads per row, bf16 store ----
  const int row = tid >> 2;
  const int sub = tid & 3;
  float s = 0.f, ss = 0.f;
  for (int c = sub * 32; c < sub * 32 + 32; ++c) {
    const float v = sOut[row * 128 + c];
    s += v; ss += v * v;
  }
  sRed[row * 8 + sub * 2]     = s;
  sRed[row * 8 + sub * 2 + 1] = ss;
  __syncthreads();
  float S = 0.f, SS = 0.f;
#pragma unroll
  for (int q = 0; q < 4; ++q) { S += sRed[row * 8 + 2 * q]; SS += sRed[row * 8 + 2 * q + 1]; }
  const float mean = S * (1.f / 128.f);
  const float var  = SS * (1.f / 128.f) - mean * mean;
  const float inv  = rsqrtf(var + LN_EPS);
  const int gm = m0 + row;
  if (gm < M) {
    const __bf16* rp = resid + (size_t)gm * 128;
    __bf16* op = out + (size_t)gm * 128;
    for (int c = sub * 32; c < sub * 32 + 32; ++c) {
      op[c] = (__bf16)((sOut[row * 128 + c] - mean) * inv * lng[c] + lnb[c] +
                       (float)rp[c]);
    }
  }
}

// =====================================================================
// Weight packer: fp32 (K x 128) -> fragment-ordered bf16.
// Packs the 4 weight matrices of one MGN block in a single launch.
// Segment offsets: eW1 384x128 @0, eW2 128x128 @49152,
//                  nW1 256x128 @65536, nW2 128x128 @98304. Total 114688.
// =====================================================================
__global__ __launch_bounds__(256) void pack_block_kernel(
    const float* __restrict__ eW1, const float* __restrict__ eW2,
    const float* __restrict__ nW1, const float* __restrict__ nW2,
    __bf16* __restrict__ out) {
  int idx = blockIdx.x * blockDim.x + threadIdx.x;
  if (idx >= 114688) return;
  const float* W;
  __bf16* o;
  int local = idx;
  if (local < 49152)       { W = eW1; o = out; }
  else if (local < 65536)  { W = eW2; o = out + 49152; local -= 49152; }
  else if (local < 98304)  { W = nW1; o = out + 65536; local -= 65536; }
  else                     { W = nW2; o = out + 98304; local -= 98304; }
  const int e    = local & 15;
  const int lane = (local >> 4) & 31;
  const int nt   = (local >> 9) & 7;
  const int ks   = local >> 12;
  const int k = ks * 32 + (e >> 3) * 16 + (lane >> 4) * 8 + (e & 7);
  const int n = nt * 16 + (lane & 15);
  o[local] = (__bf16)W[(size_t)k * 128 + n];
}

// =====================================================================
// Encoder MLP: din=3 -> 128 -> relu -> 128 -> LN. bf16 output.
// =====================================================================
__global__ __launch_bounds__(128) void mlp_enc_kernel(
    const float* __restrict__ x,
    const float* __restrict__ W1, const float* __restrict__ b1,
    const float* __restrict__ W2, const float* __restrict__ b2,
    const float* __restrict__ lng, const float* __restrict__ lnb,
    __bf16* __restrict__ out, int n) {
  __shared__ float h[128];
  __shared__ float rs[128], rq[128];
  const int i = blockIdx.x;
  if (i >= n) return;
  const int j = threadIdx.x;
  const float x0 = x[i * 3], x1 = x[i * 3 + 1], x2 = x[i * 3 + 2];
  h[j] = fmaxf(x0 * W1[j] + x1 * W1[128 + j] + x2 * W1[256 + j] + b1[j], 0.f);
  __syncthreads();
  float o = b2[j];
  for (int k = 0; k < 128; ++k) o += h[k] * W2[k * 128 + j];
  rs[j] = o; rq[j] = o * o;
  __syncthreads();
  for (int stp = 64; stp > 0; stp >>= 1) {
    if (j < stp) { rs[j] += rs[j + stp]; rq[j] += rq[j + stp]; }
    __syncthreads();
  }
  const float mean = rs[0] * (1.f / 128.f);
  const float var  = rq[0] * (1.f / 128.f) - mean * mean;
  const float inv  = rsqrtf(var + LN_EPS);
  out[(size_t)i * 128 + j] = (__bf16)((o - mean) * inv * lng[j] + lnb[j]);
}

// =====================================================================
// Decoder MLP: 128(bf16) -> 128 -> relu -> 3(fp32), optional LN over 3.
// =====================================================================
__global__ __launch_bounds__(128) void mlp_dec_kernel(
    const __bf16* __restrict__ x,
    const float* __restrict__ W1, const float* __restrict__ b1,
    const float* __restrict__ W2, const float* __restrict__ b2,
    const float* __restrict__ g3, const float* __restrict__ b3,
    float* __restrict__ out, int n) {
  __shared__ float xi[128], h[128], o3[3];
  const int i = blockIdx.x;
  if (i >= n) return;
  const int j = threadIdx.x;
  xi[j] = (float)x[(size_t)i * 128 + j];
  __syncthreads();
  float acc = b1[j];
  for (int k = 0; k < 128; ++k) acc += xi[k] * W1[k * 128 + j];
  h[j] = fmaxf(acc, 0.f);
  __syncthreads();
  if (j < 3) {
    float o = b2[j];
    for (int k = 0; k < 128; ++k) o += h[k] * W2[k * 3 + j];
    o3[j] = o;
  }
  __syncthreads();
  if (g3 != nullptr) {
    if (j == 0) {
      const float m = (o3[0] + o3[1] + o3[2]) * (1.f / 3.f);
      float v = 0.f;
      for (int c = 0; c < 3; ++c) v += (o3[c] - m) * (o3[c] - m);
      v *= (1.f / 3.f);
      const float inv = rsqrtf(v + LN_EPS);
      for (int c = 0; c < 3; ++c) out[i * 3 + c] = (o3[c] - m) * inv * g3[c] + b3[c];
    }
  } else if (j < 3) {
    out[i * 3 + j] = o3[j];
  }
}

// =====================================================================
// Aux kernels
// =====================================================================
__global__ void zero_kernel(float* __restrict__ p, int n) {
  const int i = blockIdx.x * blockDim.x + threadIdx.x;
  if (i < n) p[i] = 0.f;
}

__global__ void count_deg_kernel(const int* __restrict__ dst, int E, float* __restrict__ deg) {
  const int e = blockIdx.x * blockDim.x + threadIdx.x;
  if (e < E) atomicAdd(&deg[dst[e]], 1.f);
}

__global__ void invdeg_kernel(float* __restrict__ deg, int n) {
  const int i = blockIdx.x * blockDim.x + threadIdx.x;
  if (i < n) deg[i] = 1.f / fmaxf(deg[i], 1.f);
}

__global__ void scatter_add_kernel(const __bf16* __restrict__ ef, const int* __restrict__ dst,
                                   int E, float* __restrict__ agg) {
  const int i = blockIdx.x * blockDim.x + threadIdx.x;
  if (i >= E * 128) return;
  const int e = i >> 7, c = i & 127;
  atomicAdd(&agg[(size_t)dst[e] * 128 + c], (float)ef[(size_t)e * 128 + c]);
}

// KNN(k=3) interpolation: one thread per target point, D=3 features.
__global__ void knn_kernel(const float* __restrict__ feat, const float* __restrict__ pos_src,
                           const float* __restrict__ pos_tgt, float* __restrict__ out,
                           int ns, int nt) {
  const int t = blockIdx.x * blockDim.x + threadIdx.x;
  if (t >= nt) return;
  const float px = pos_tgt[t * 3], py = pos_tgt[t * 3 + 1], pz = pos_tgt[t * 3 + 2];
  float bd0 = 1e30f, bd1 = 1e30f, bd2 = 1e30f;
  int bi0 = 0, bi1 = 0, bi2 = 0;
  for (int s = 0; s < ns; ++s) {
    const float dx = px - pos_src[s * 3];
    const float dy = py - pos_src[s * 3 + 1];
    const float dz = pz - pos_src[s * 3 + 2];
    const float d = dx * dx + dy * dy + dz * dz;
    if (d < bd0)      { bd2 = bd1; bi2 = bi1; bd1 = bd0; bi1 = bi0; bd0 = d; bi0 = s; }
    else if (d < bd1) { bd2 = bd1; bi2 = bi1; bd1 = d; bi1 = s; }
    else if (d < bd2) { bd2 = d; bi2 = s; }
  }
  const float w0 = 1.f / fmaxf(bd0, 1e-16f);
  const float w1 = 1.f / fmaxf(bd1, 1e-16f);
  const float w2 = 1.f / fmaxf(bd2, 1e-16f);
  const float iw = 1.f / (w0 + w1 + w2);
  for (int c = 0; c < 3; ++c) {
    out[t * 3 + c] = (w0 * feat[bi0 * 3 + c] + w1 * feat[bi1 * 3 + c] +
                      w2 * feat[bi2 * 3 + c]) * iw;
  }
}

// =====================================================================
// Host side: pytree unpack + pipeline
// =====================================================================
struct Mlp { const float *W1, *W2, *b1, *b2, *lnb, *lng; };

static inline Mlp mk_mlp(void* const* d_in, int base, bool ln) {
  Mlp m;
  m.W1 = (const float*)d_in[base];
  m.W2 = (const float*)d_in[base + 1];
  m.b1 = (const float*)d_in[base + 2];
  m.b2 = (const float*)d_in[base + 3];
  m.lnb = ln ? (const float*)d_in[base + 4] : nullptr;
  m.lng = ln ? (const float*)d_in[base + 5] : nullptr;
  return m;
}

static const int PACK_BLK = 114688;  // packed bf16 elems per MGN block

static void run_mgn(void* const* d_in, int pbase,
                    const float* node_in3,   // fp32 din=3 node input
                    const float* efeat,
                    const int* src, const int* dst, const float* invdeg,
                    int N, int E,
                    __bf16* nfA, __bf16* nfB, __bf16* efA, __bf16* efB,
                    float* agg, __bf16* wpack,
                    const float* ln3g, const float* ln3b, float* lat,
                    hipStream_t stream) {
  // pytree leaf offsets inside one MGN param set (alphabetical keys):
  // blocks[0..14]{edge(6),node(6)}=180, edge_enc(6)@180, node_dec(4)@186, node_enc(6)@190
  const Mlp node_enc = mk_mlp(d_in, pbase + 190, true);
  const Mlp edge_enc = mk_mlp(d_in, pbase + 180, true);
  const Mlp node_dec = mk_mlp(d_in, pbase + 186, false);

  // pack all block weights to fragment-ordered bf16
  for (int blk = 0; blk < 15; ++blk) {
    const Mlp me = mk_mlp(d_in, pbase + blk * 12, true);
    const Mlp mn = mk_mlp(d_in, pbase + blk * 12 + 6, true);
    pack_block_kernel<<<(PACK_BLK + 255) / 256, 256, 0, stream>>>(
        me.W1, me.W2, mn.W1, mn.W2, wpack + (size_t)blk * PACK_BLK);
  }

  mlp_enc_kernel<<<N, 128, 0, stream>>>(node_in3, node_enc.W1, node_enc.b1,
      node_enc.W2, node_enc.b2, node_enc.lng, node_enc.lnb, nfA, N);
  mlp_enc_kernel<<<E, 128, 0, stream>>>(efeat, edge_enc.W1, edge_enc.b1,
      edge_enc.W2, edge_enc.b2, edge_enc.lng, edge_enc.lnb, efA, E);

  __bf16* nfs = nfA; __bf16* nfd = nfB;
  __bf16* efs = efA; __bf16* efd = efB;
  for (int blk = 0; blk < 15; ++blk) {
    const Mlp me = mk_mlp(d_in, pbase + blk * 12, true);
    const Mlp mn = mk_mlp(d_in, pbase + blk * 12 + 6, true);
    const __bf16* wb = wpack + (size_t)blk * PACK_BLK;
    // edge block
    mgn_block_kernel<384><<<(E + 63) / 64, 256, 0, stream>>>(
        efs, nullptr, nullptr, nfs, src, dst,
        wb, me.b1, wb + 49152, me.b2, me.lng, me.lnb, efs, efd, E);
    // mean-aggregate incoming (updated) edge features
    zero_kernel<<<(N * 128 + 255) / 256, 256, 0, stream>>>(agg, N * 128);
    scatter_add_kernel<<<(E * 128 + 255) / 256, 256, 0, stream>>>(efd, dst, E, agg);
    // node block
    mgn_block_kernel<256><<<(N + 63) / 64, 256, 0, stream>>>(
        nullptr, agg, invdeg, nfs, nullptr, nullptr,
        wb + 65536, mn.b1, wb + 98304, mn.b2, mn.lng, mn.lnb, nfs, nfd, N);
    __bf16* t;
    t = efs; efs = efd; efd = t;
    t = nfs; nfs = nfd; nfd = t;
  }
  mlp_dec_kernel<<<N, 128, 0, stream>>>(nfs, node_dec.W1, node_dec.b1,
      node_dec.W2, node_dec.b2, ln3g, ln3b, lat, N);
}

extern "C" void kernel_launch(void* const* d_in, const int* in_sizes, int n_in,
                              void* d_out, int out_size, void* d_ws, size_t ws_size,
                              hipStream_t stream) {
  const int N  = in_sizes[0] / 3;
  const int E  = in_sizes[1] / 3;
  const int NP = in_sizes[4] / 3;

  const float* nfeat = (const float*)d_in[0];
  const float* efeat = (const float*)d_in[1];
  const int*   eidx  = (const int*)d_in[2];
  const int*   src   = eidx;
  const int*   dst   = eidx + E;
  const float* pmesh = (const float*)d_in[3];
  const float* ppiv  = (const float*)d_in[4];
  const int encBase = 5;
  const int decBase = 5 + 196;
  const float* piv_b = (const float*)d_in[5 + 2 * 196];      // piv_ln["b"]
  const float* piv_g = (const float*)d_in[5 + 2 * 196 + 1];  // piv_ln["g"]

  char* cur = (char*)d_ws;
  auto alloc = [&](size_t bytes) -> void* {
    void* p = (void*)cur;
    cur += (bytes + 63) & ~(size_t)63;
    return p;
  };
  __bf16* nfA    = (__bf16*)alloc((size_t)N * 128 * 2);
  __bf16* nfB    = (__bf16*)alloc((size_t)N * 128 * 2);
  __bf16* efA    = (__bf16*)alloc((size_t)E * 128 * 2);
  __bf16* efB    = (__bf16*)alloc((size_t)E * 128 * 2);
  float*  agg    = (float*)alloc((size_t)N * 128 * 4);
  float*  invdeg = (float*)alloc((size_t)N * 4);
  float*  lat    = (float*)alloc((size_t)N * 3 * 4);
  float*  z      = (float*)alloc((size_t)NP * 3 * 4);
  float*  y      = (float*)alloc((size_t)N * 3 * 4);
  __bf16* wpackE = (__bf16*)alloc((size_t)15 * PACK_BLK * 2);
  __bf16* wpackD = (__bf16*)alloc((size_t)15 * PACK_BLK * 2);

  // degree (shared by both MGNs)
  zero_kernel<<<(N + 255) / 256, 256, 0, stream>>>(invdeg, N);
  count_deg_kernel<<<(E + 255) / 256, 256, 0, stream>>>(dst, E, invdeg);
  invdeg_kernel<<<(N + 255) / 256, 256, 0, stream>>>(invdeg, N);

  // encoder MGN -> piv LN -> knn to pivotal
  run_mgn(d_in, encBase, nfeat, efeat, src, dst, invdeg, N, E,
          nfA, nfB, efA, efB, agg, wpackE, piv_g, piv_b, lat, stream);
  knn_kernel<<<(NP + 127) / 128, 128, 0, stream>>>(lat, pmesh, ppiv, z, N, NP);

  // knn back to mesh -> decoder MGN -> d_out
  knn_kernel<<<(N + 127) / 128, 128, 0, stream>>>(z, ppiv, pmesh, y, NP, N);
  run_mgn(d_in, decBase, y, efeat, src, dst, invdeg, N, E,
          nfA, nfB, efA, efB, agg, wpackD, nullptr, nullptr, (float*)d_out, stream);
}
